// Temper_73332271612749
// MI455X (gfx1250) — matrix operand order repaired and checked by reading, not compiled, not run on hardware
//
#include <hip/hip_runtime.h>

typedef __attribute__((ext_vector_type(16))) __bf16 v16bf;
typedef __attribute__((ext_vector_type(8)))  __bf16 v8bf;
typedef __attribute__((ext_vector_type(8)))  float  v8f;
typedef __attribute__((ext_vector_type(4)))  float  v4f;

union Frag {
    v16bf v;
    v8bf  h[2];
    unsigned int w[8];
};

#define NROW  131072
#define DIM   256
#define HID   256
#define MSLOT 50
#define ROWS  128
#define XPLD  260   // f32 xp row stride (floats)
#define ABLD  528   // interleaved hi/lo row stride (bf16 elems): [hi 264 | lo 264]
#define NBLK  (NROW / ROWS)

// packed split: two fp32 -> (2x bf16 hi, 2x bf16 lo) using v_perm_b32.
// hi = round-half-up(f) to bf16 (<=1 ulp), lo = trunc(f - hi) -> ~2^-17 combined.
__device__ __forceinline__ void split_pair(float f0, float f1,
                                           unsigned int& hi2, unsigned int& lo2) {
    unsigned int u0 = __float_as_uint(f0), u1 = __float_as_uint(f1);
    unsigned int r0 = u0 + 0x8000u, r1 = u1 + 0x8000u;
    hi2 = __builtin_amdgcn_perm(r1, r0, 0x07060302u);           // {r1.hi16, r0.hi16}
    float b0 = __uint_as_float(r0 & 0xffff0000u);
    float b1 = __uint_as_float(r1 & 0xffff0000u);
    unsigned int d0 = __float_as_uint(f0 - b0);
    unsigned int d1 = __float_as_uint(f1 - b1);
    lo2 = __builtin_amdgcn_perm(d1, d0, 0x07060302u);           // truncated residuals
}

__device__ __forceinline__ void split_b16(const float* wp, Frag& Bh, Frag& Bl) {
    v4f w0 = *(const v4f*)(wp);
    v4f w1 = *(const v4f*)(wp + 4);
    v4f w2 = *(const v4f*)(wp + 16);
    v4f w3 = *(const v4f*)(wp + 20);
    split_pair(w0[0], w0[1], Bh.w[0], Bl.w[0]);
    split_pair(w0[2], w0[3], Bh.w[1], Bl.w[1]);
    split_pair(w1[0], w1[1], Bh.w[2], Bl.w[2]);
    split_pair(w1[2], w1[3], Bh.w[3], Bl.w[3]);
    split_pair(w2[0], w2[1], Bh.w[4], Bl.w[4]);
    split_pair(w2[2], w2[3], Bh.w[5], Bl.w[5]);
    split_pair(w3[0], w3[1], Bh.w[6], Bl.w[6]);
    split_pair(w3[2], w3[3], Bh.w[7], Bl.w[7]);
}

__global__ __launch_bounds__(256) void temper_main(
    const float* __restrict__ x,
    const float* __restrict__ mem,
    const float* __restrict__ W_in,
    const float* __restrict__ b_in,
    const float* __restrict__ W_ops,
    const float* __restrict__ b_ops,
    const float* __restrict__ routing,
    float* __restrict__ out,
    float* __restrict__ part1,
    float* __restrict__ part2)
{
    extern __shared__ char smem[];
    unsigned short* sh_ab = (unsigned short*)smem;        // ROWS*ABLD bf16 (hi|lo interleaved per row)
    unsigned int*   sab32 = (unsigned int*)smem;          // 32-bit view for packed stores
    float*          xp    = (float*)(sh_ab + ROWS * ABLD);// ROWS*XPLD f32 (pre-relu x_proj)
    float*          red   = xp + ROWS * XPLD;             // 512 f32

    const int tid  = threadIdx.x;
    const int wv   = tid >> 5;
    const int lane = tid & 31;
    const int grp  = lane >> 4;
    const int l16  = lane & 15;
    const long rowBase = (long)blockIdx.x * ROWS;

    // ---------- softmax of routing logits (uniform) ----------
    float lg0 = routing[0], lg1 = routing[1], lg2 = routing[2];
    float mxl = fmaxf(lg0, fmaxf(lg1, lg2));
    float e0 = expf(lg0 - mxl), e1 = expf(lg1 - mxl), e2 = expf(lg2 - mxl);
    float esum = e0 + e1 + e2;
    float wop[3] = { e0 / esum, e1 / esum, e2 / esum };

    // ---------- phase 0: x tile -> LDS bf16 hi/lo (packed stores) ----------
    {
        const float* xt = x + rowBase * DIM;
        #pragma unroll
        for (int i = 0; i < 32; ++i) {
            int e = i * 1024 + tid * 4;
            int r = e >> 8, c = e & 255;
            v4f xv = *(const v4f*)(xt + e);
            unsigned int h0, l0, h1, l1;
            split_pair(xv[0], xv[1], h0, l0);
            split_pair(xv[2], xv[3], h1, l1);
            int wbase = r * (ABLD / 2) + (c >> 1);
            sab32[wbase]           = h0;
            sab32[wbase + 1]       = h1;
            sab32[wbase + 132]     = l0;      // lo half starts 264 elems (132 words) in
            sab32[wbase + 132 + 1] = l1;
        }
    }
    __syncthreads();

    v8f zero = {};
    // per-lane invariant A-fragment bases; the +4-row-tile displacement is made
    // opaque so the compiler keeps a second base register instead of folding it
    // into >64KB DS-offset immediates (which forced per-iteration v_adds).
    const unsigned short* ab0 = sh_ab + l16 * ABLD + grp * 8;
    int far = 4 * 16 * ABLD;
    asm volatile("" : "+v"(far));
    const unsigned short* ab4 = ab0 + far;

    // ---------- GEMM1: x_proj = x @ W_in^T + b_in ----------
    for (int c = 0; c < 2; ++c) {
        const int colBase = wv * 32 + c * 16;
        v8f Acc[8];
        #pragma unroll
        for (int rt = 0; rt < 8; ++rt) Acc[rt] = zero;

        for (int ks = 0; ks < 8; ++ks) {
            const int kb = ks * 32;
            Frag Bh, Bl;
            split_b16(W_in + (colBase + l16) * DIM + kb + grp * 8, Bh, Bl);
            #pragma unroll
            for (int rt = 0; rt < 8; ++rt) {
                Frag Ah, Al;
                const unsigned short* ab =
                    ((rt < 4) ? ab0 : ab4) + (rt & 3) * 16 * ABLD + kb;  // const DS offsets
                Ah.h[0] = *(const v8bf*)(ab);
                Ah.h[1] = *(const v8bf*)(ab + 16);
                Al.h[0] = *(const v8bf*)(ab + 264);
                Al.h[1] = *(const v8bf*)(ab + 280);
                Acc[rt] = __builtin_amdgcn_wmma_f32_16x16x32_bf16(false, Ah.v, false, Bh.v, (short)0, Acc[rt], false, false);
                Acc[rt] = __builtin_amdgcn_wmma_f32_16x16x32_bf16(false, Ah.v, false, Bl.v, (short)0, Acc[rt], false, false);
                Acc[rt] = __builtin_amdgcn_wmma_f32_16x16x32_bf16(false, Al.v, false, Bh.v, (short)0, Acc[rt], false, false);
            }
        }
        const float bi = b_in[colBase + l16];
        #pragma unroll
        for (int rt = 0; rt < 8; ++rt) {
            #pragma unroll
            for (int vv = 0; vv < 8; ++vv) {
                int rl = rt * 16 + grp * 8 + vv;   // C layout: m = vv + 8*grp, n = l16
                xp[rl * XPLD + colBase + l16] = Acc[rt][vv] + bi;
            }
        }
    }
    __syncthreads();

    // ---------- phase 2: relu(x_proj) -> bf16 hi/lo (overwrites dead x tile) ----------
    {
        #pragma unroll
        for (int i = 0; i < 32; ++i) {
            int e = i * 1024 + tid * 4;
            int r = e >> 8, c = e & 255;
            v4f xv = *(const v4f*)(xp + r * XPLD + c);
            unsigned int h0, l0, h1, l1;
            split_pair(fmaxf(xv[0], 0.f), fmaxf(xv[1], 0.f), h0, l0);
            split_pair(fmaxf(xv[2], 0.f), fmaxf(xv[3], 0.f), h1, l1);
            int wbase = r * (ABLD / 2) + (c >> 1);
            sab32[wbase]           = h0;
            sab32[wbase + 1]       = h1;
            sab32[wbase + 132]     = l0;
            sab32[wbase + 132 + 1] = l1;
        }
    }

    // ---------- distances to memory bank: block-local sum(d), sum(d^2) ----------
    {
        int row = tid >> 1;
        int mb  = (tid & 1) * 25;
        const float* xr = xp + row * XPLD;
        float s1 = 0.f, s2 = 0.f;
        for (int m = 0; m < 25; ++m) {
            const float* mp = mem + (mb + m) * HID;
            float acc = 0.f;
            #pragma unroll 8
            for (int h4 = 0; h4 < 64; ++h4) {
                v4f a = *(const v4f*)(xr + h4 * 4);
                v4f b = *(const v4f*)(mp + h4 * 4);
                v4f d = a - b;
                acc += d[0] * d[0] + d[1] * d[1] + d[2] * d[2] + d[3] * d[3];
            }
            s1 += sqrtf(acc);
            s2 += acc;
        }
        red[tid] = s1;
        red[256 + tid] = s2;
    }
    __syncthreads();
    for (int s = 128; s > 0; s >>= 1) {   // barriers also publish phase-2 LDS writes
        if (tid < s) { red[tid] += red[tid + s]; red[256 + tid] += red[256 + tid + s]; }
        __syncthreads();
    }
    if (tid == 0) { part1[blockIdx.x] = red[0]; part2[blockIdx.x] = red[256]; }

    // ---------- GEMM2..4: out = sum_o w_o * relu(relu(xp) @ W_ops[o]^T + b_ops[o]) ----------
    for (int c = 0; c < 2; ++c) {
        const int colBase = wv * 32 + c * 16;
        v8f Out[8];
        #pragma unroll
        for (int rt = 0; rt < 8; ++rt) Out[rt] = zero;

        for (int o = 0; o < 3; ++o) {
            const float* W = W_ops + o * (HID * HID);
            v8f Acc[8];
            #pragma unroll
            for (int rt = 0; rt < 8; ++rt) Acc[rt] = zero;

            for (int ks = 0; ks < 8; ++ks) {
                const int kb = ks * 32;
                Frag Bh, Bl;
                split_b16(W + (colBase + l16) * HID + kb + grp * 8, Bh, Bl);
                #pragma unroll
                for (int rt = 0; rt < 8; ++rt) {
                    Frag Ah, Al;
                    const unsigned short* ab =
                        ((rt < 4) ? ab0 : ab4) + (rt & 3) * 16 * ABLD + kb;
                    Ah.h[0] = *(const v8bf*)(ab);
                    Ah.h[1] = *(const v8bf*)(ab + 16);
                    Al.h[0] = *(const v8bf*)(ab + 264);
                    Al.h[1] = *(const v8bf*)(ab + 280);
                    Acc[rt] = __builtin_amdgcn_wmma_f32_16x16x32_bf16(false, Ah.v, false, Bh.v, (short)0, Acc[rt], false, false);
                    Acc[rt] = __builtin_amdgcn_wmma_f32_16x16x32_bf16(false, Ah.v, false, Bl.v, (short)0, Acc[rt], false, false);
                    Acc[rt] = __builtin_amdgcn_wmma_f32_16x16x32_bf16(false, Al.v, false, Bh.v, (short)0, Acc[rt], false, false);
                }
            }
            const float wo = wop[o];
            const float bo = b_ops[o * HID + colBase + l16];
            #pragma unroll
            for (int rt = 0; rt < 8; ++rt) {
                #pragma unroll
                for (int vv = 0; vv < 8; ++vv)
                    Out[rt][vv] += wo * fmaxf(Acc[rt][vv] + bo, 0.f);
            }
        }
        #pragma unroll
        for (int rt = 0; rt < 8; ++rt) {
            #pragma unroll
            for (int vv = 0; vv < 8; ++vv) {
                long row = rowBase + rt * 16 + grp * 8 + vv;
                out[row * HID + colBase + l16] = Out[rt][vv];
            }
        }
    }
}

// ---------------- finalization: novelty / conflict / controller ----------------
__global__ void temper_ctrl(
    const float* __restrict__ routing,
    const float* __restrict__ Wc1, const float* __restrict__ bc1,
    const float* __restrict__ Wc2, const float* __restrict__ bc2,
    const float* __restrict__ part1, const float* __restrict__ part2,
    float* __restrict__ out)
{
    if (threadIdx.x != 0 || blockIdx.x != 0) return;
    double S1 = 0.0, S2 = 0.0;
    for (int b = 0; b < NBLK; ++b) { S1 += (double)part1[b]; S2 += (double)part2[b]; }
    const double NM = (double)NROW * (double)MSLOT;
    double var = (S2 - S1 * S1 / NM) / (NM - 1.0);
    if (var < 0.0) var = 0.0;
    float novelty = fminf(1.5f, (float)sqrt(var));

    float lg0 = routing[0], lg1 = routing[1], lg2 = routing[2];
    float mxl = fmaxf(lg0, fmaxf(lg1, lg2));
    float e0 = expf(lg0 - mxl), e1 = expf(lg1 - mxl), e2 = expf(lg2 - mxl);
    float es = e0 + e1 + e2;
    float w0 = e0 / es, w1 = e1 / es, w2 = e2 / es;
    float mw = (w0 + w1 + w2) * (1.0f / 3.0f);
    float conflict = sqrtf(((w0 - mw) * (w0 - mw) + (w1 - mw) * (w1 - mw) + (w2 - mw) * (w2 - mw)) * 0.5f);

    float stats[5] = { novelty, conflict, 1.0f, 0.0f, 0.001f };
    float hdn[8];
    for (int j = 0; j < 8; ++j) {
        float t = bc1[j];
        for (int i = 0; i < 5; ++i) t += stats[i] * Wc1[j * 5 + i];
        hdn[j] = fmaxf(t, 0.f);
    }
    for (int k = 0; k < 2; ++k) {
        float t = bc2[k];
        for (int j = 0; j < 8; ++j) t += hdn[j] * Wc2[k * 8 + j];
        out[(long)NROW * HID + k] = 1.0f / (1.0f + expf(-t));
    }
}

extern "C" void kernel_launch(void* const* d_in, const int* in_sizes, int n_in,
                              void* d_out, int out_size, void* d_ws, size_t ws_size,
                              hipStream_t stream) {
    (void)in_sizes; (void)n_in; (void)out_size; (void)ws_size;
    const float* x       = (const float*)d_in[0];
    const float* memory  = (const float*)d_in[1];
    const float* W_in    = (const float*)d_in[2];
    const float* b_in    = (const float*)d_in[3];
    const float* W_ops   = (const float*)d_in[4];
    const float* b_ops   = (const float*)d_in[5];
    const float* routing = (const float*)d_in[6];
    const float* Wc1     = (const float*)d_in[7];
    const float* bc1     = (const float*)d_in[8];
    const float* Wc2     = (const float*)d_in[9];
    const float* bc2     = (const float*)d_in[10];
    float* out = (float*)d_out;
    float* part1 = (float*)d_ws;
    float* part2 = part1 + NBLK;

    const int dynLds = ROWS * ABLD * 2      // interleaved hi/lo bf16
                     + ROWS * XPLD * 4      // xp f32
                     + 512 * 4;             // reduction
    hipFuncSetAttribute((const void*)temper_main,
                        hipFuncAttributeMaxDynamicSharedMemorySize, dynLds);

    temper_main<<<NBLK, 256, dynLds, stream>>>(x, memory, W_in, b_in, W_ops, b_ops,
                                               routing, out, part1, part2);
    temper_ctrl<<<1, 1, 0, stream>>>(routing, Wc1, bc1, Wc2, bc2, part1, part2, out);
}
